// TopKSearchDecoder_55697135894915
// MI455X (gfx1250) — compile-verified, matrix operand order checked by reading, not run on hardware
//
#include <hip/hip_runtime.h>
#include <hip/hip_bf16.h>

#define SS 256
#define HH 512
#define G3 1536
#define VV 50257
#define VP 50304     // V padded to multiple of 64 (4 tiles of 16 per block)
#define KB 8
#define TT 64
#define SOS_TOK 1

typedef __attribute__((ext_vector_type(2)))  float  v2f;
typedef __attribute__((ext_vector_type(8)))  float  v8f;
typedef __attribute__((ext_vector_type(8)))  __bf16 v8bf;
typedef __attribute__((ext_vector_type(16))) __bf16 v16bf;

typedef __attribute__((address_space(3))) __bf16 lds_bf16;

__device__ __forceinline__ float sigf(float x) { return 1.0f / (1.0f + expf(-x)); }

// ---------------------------------------------------------------------------
// Generic fp32 WMMA GEMM: C[M][N] = A[M][K] @ W[N][K]^T + bias[N]
// one wave per 16x16 output tile; v_wmma_f32_16x16x4_f32, K-loop step 4.
// revA: A row index reversed (for backward-direction encoder input gates).
// ---------------------------------------------------------------------------
__global__ void gemm_f32_wmma(const float* __restrict__ A, int lda, int arows, int revA,
                              const float* __restrict__ W, int Kdim,
                              const float* __restrict__ bias,
                              float* __restrict__ C, int ldc)
{
  const int lane = threadIdx.x & 31;
  const int mn   = lane & 15;
  const int half = lane >> 4;
  const int n0 = blockIdx.x * 16;
  const int m0 = blockIdx.y * 16;
  int arow = m0 + mn;
  if (revA) arow = arows - 1 - arow;
  const float* Ap = A + (size_t)arow * lda;
  const float* Wp = W + (size_t)(n0 + mn) * Kdim;
  v8f acc = {0.f,0.f,0.f,0.f,0.f,0.f,0.f,0.f};
  for (int kt = 0; kt < Kdim; kt += 4) {
    v2f a = *reinterpret_cast<const v2f*>(Ap + kt + 2*half);
    v2f b = *reinterpret_cast<const v2f*>(Wp + kt + 2*half);
    acc = __builtin_amdgcn_wmma_f32_16x16x4_f32(false, a, false, b, (short)0, acc,
                                                false, false);
  }
  const int col = n0 + mn;
  const float bv = bias ? bias[col] : 0.f;
#pragma unroll
  for (int r = 0; r < 8; ++r) {
    const int row = m0 + r + 8*half;
    C[(size_t)row*ldc + col] = acc[r] + bv;
  }
}

// ---------------------------------------------------------------------------
// bf16 WMMA GEMM for the output projection: C[16][N] = A[16][512] @ W[N][512]^T + b
// 128-thread blocks (4 waves), 4 N-tiles per block. The 16KB A matrix is staged
// into LDS once per block with global_load_async_to_lds_b128 (ASYNCcnt), then
// each wave streams its W tiles from L2 (Wout bf16 = 51.5MB, L2-resident) and
// reads A fragments from LDS. v_wmma_f32_16x16x32_bf16.
// ---------------------------------------------------------------------------
__global__ void gemm_bf16_wmma(const __bf16* __restrict__ A,   // 16 x 512
                               const __bf16* __restrict__ W,   // N x 512
                               const float* __restrict__ bias, int nbias,
                               float* __restrict__ C, int ldc)
{
  __shared__ __bf16 sA[16*HH];                  // 16384 bytes
  const int tid = threadIdx.x;                  // 128
  {
    lds_bf16* s3 = (lds_bf16*)sA;
    // 16KB / (128 threads * 16B) = 8 passes of b128 async copies
#pragma unroll
    for (int pass = 0; pass < 8; ++pass) {
      const int eoff = (pass*128 + tid) * 8;    // 8 bf16 = 16 bytes per lane
      asm volatile("global_load_async_to_lds_b128 %0, %1, off"
                   :: "v"(s3 + eoff), "v"(A + eoff) : "memory");
    }
    asm volatile("s_wait_asynccnt 0x0" ::: "memory");
  }
  __syncthreads();

  const int wave = tid >> 5;
  const int lane = tid & 31;
  const int mn   = lane & 15;
  const int half = lane >> 4;
  const int n0 = (blockIdx.x * 4 + wave) * 16;
  const __bf16* Wp = W + (size_t)(n0 + mn) * HH;
  const __bf16* Ap = sA + mn * HH;              // LDS row for this lane's M
  v8f acc = {0.f,0.f,0.f,0.f,0.f,0.f,0.f,0.f};
  for (int kt = 0; kt < HH; kt += 32) {
    if (kt + 32 < HH) __builtin_prefetch(Wp + kt + 32, 0, 1);  // global_prefetch_b8
    v8bf alo = *reinterpret_cast<const v8bf*>(Ap + kt + 8*half);       // ds_load_b128
    v8bf ahi = *reinterpret_cast<const v8bf*>(Ap + kt + 16 + 8*half);
    v8bf blo = *reinterpret_cast<const v8bf*>(Wp + kt + 8*half);
    v8bf bhi = *reinterpret_cast<const v8bf*>(Wp + kt + 16 + 8*half);
    v16bf a, b;
#pragma unroll
    for (int i = 0; i < 8; ++i) { a[i]=alo[i]; a[8+i]=ahi[i]; b[i]=blo[i]; b[8+i]=bhi[i]; }
    acc = __builtin_amdgcn_wmma_f32_16x16x32_bf16(false, a, false, b, (short)0, acc,
                                                  false, false);
  }
  const int col = n0 + mn;
  const float bv = (bias && col < nbias) ? bias[col] : 0.f;
#pragma unroll
  for (int r = 0; r < 8; ++r)
    C[(size_t)(r + 8*half)*ldc + col] = acc[r] + bv;
}

// ---------------------------------------------------------------------------
// prep kernels
// ---------------------------------------------------------------------------
__global__ void embed_gather_kernel(const int* __restrict__ seq,
                                    const float* __restrict__ emb,
                                    float* __restrict__ EMB)
{
  const int s = blockIdx.x, i = threadIdx.x;
  EMB[(size_t)s*HH + i] = emb[(size_t)seq[s]*HH + i];
}

__global__ void wout_convert_kernel(const float* __restrict__ Wout, __bf16* __restrict__ WB)
{
  const size_t idx = (size_t)blockIdx.x * blockDim.x + threadIdx.x;
  if (idx >= (size_t)VP * HH) return;
  const size_t row = idx / HH;
  WB[idx] = (row < VV) ? (__bf16)Wout[idx] : (__bf16)0.f;
}

// ---------------------------------------------------------------------------
// sequential GRU over S steps (one direction); single workgroup, h in LDS.
// GX already holds x@Wih^T + bih for each step (input-gate precompute).
// forward: enc_out[t] = h ; backward: enc_out[S-1-t] += h.
// ---------------------------------------------------------------------------
__global__ void enc_dir_kernel(const float* __restrict__ GX,
                               const float* __restrict__ Whh,
                               const float* __restrict__ bhh,
                               float* __restrict__ enc_out,
                               float* __restrict__ h_final, int reverse)
{
  __shared__ float hs[HH];
  const int i = threadIdx.x;  // 512 threads
  hs[i] = 0.f;
  __syncthreads();
  for (int t = 0; t < SS; ++t) {
    float gh[3];
#pragma unroll
    for (int c = 0; c < 3; ++c) {
      const int row = c*HH + i;
      const float* wr = Whh + (size_t)row*HH;
      float s = bhh[row];
      for (int k = 0; k < HH; ++k) s += wr[k]*hs[k];
      gh[c] = s;
    }
    const float* gx = GX + (size_t)t*G3;
    const float r = sigf(gx[i]        + gh[0]);
    const float z = sigf(gx[HH + i]   + gh[1]);
    const float n = tanhf(gx[2*HH + i] + r*gh[2]);
    const float hn = (1.f - z)*n + z*hs[i];
    __syncthreads();           // everyone done reading hs
    hs[i] = hn;
    float* orow = enc_out + (size_t)(reverse ? (SS-1-t) : t) * HH;
    if (reverse) orow[i] += hn; else orow[i] = hn;
    __syncthreads();
  }
  if (!reverse) h_final[i] = hs[i];
}

// ---------------------------------------------------------------------------
// decoder-step helper kernels
// ---------------------------------------------------------------------------
__global__ void hid_init_kernel(const float* __restrict__ DECH0,
                                float* __restrict__ HID, float* __restrict__ HID2,
                                __bf16* __restrict__ CATB)
{
  const int b = blockIdx.x, i = threadIdx.x;       // 16 x 512
  HID [b*HH + i] = (b == 0) ? DECH0[i] : 0.f;
  HID2[b*HH + i] = 0.f;
  CATB[b*HH + i] = (__bf16)0.f;
}

__global__ void gather_x_kernel(const float* __restrict__ emb,
                                const int* __restrict__ last,
                                int beams, int use_sos, float* __restrict__ X)
{
  const int b = blockIdx.x, i = threadIdx.x;       // 16 x 512
  float v = 0.f;
  if (b < beams) {
    const int tok = use_sos ? SOS_TOK : last[b];
    v = emb[(size_t)tok*HH + i];
  }
  X[b*HH + i] = v;
}

__global__ void gru_combine_kernel(const float* __restrict__ GI,
                                   const float* __restrict__ GH,
                                   const float* __restrict__ HID,
                                   float* __restrict__ HID2)
{
  const int b = blockIdx.x, i = threadIdx.x;       // beams x 512
  const float xr = GI[b*G3 + i], xz = GI[b*G3 + HH + i], xn = GI[b*G3 + 2*HH + i];
  const float hr = GH[b*G3 + i], hz = GH[b*G3 + HH + i], hn = GH[b*G3 + 2*HH + i];
  const float r = sigf(xr + hr);
  const float z = sigf(xz + hz);
  const float n = tanhf(xn + r*hn);
  HID2[b*HH + i] = (1.f - z)*n + z*HID[b*HH + i];
}

__global__ void attn_kernel(const float* __restrict__ HID2,
                            const float* __restrict__ enc_out,
                            float* __restrict__ CTX)
{
  __shared__ float h2s[HH];
  __shared__ float aw[SS];
  __shared__ float red[256];
  const int b = blockIdx.x, tid = threadIdx.x;     // beams x 256
  h2s[tid]       = HID2[b*HH + tid];
  h2s[tid + 256] = HID2[b*HH + tid + 256];
  __syncthreads();
  const float* er = enc_out + (size_t)tid*HH;
  float s = 0.f;
  for (int k = 0; k < HH; ++k) s += h2s[k]*er[k];
  aw[tid] = s;
  red[tid] = s;
  for (int off = 128; off > 0; off >>= 1) { __syncthreads(); if (tid < off) red[tid] = fmaxf(red[tid], red[tid+off]); }
  __syncthreads();
  const float m = red[0];
  __syncthreads();
  const float e = expf(aw[tid] - m);
  red[tid] = e;
  for (int off = 128; off > 0; off >>= 1) { __syncthreads(); if (tid < off) red[tid] += red[tid+off]; }
  __syncthreads();
  const float inv = 1.f / red[0];
  __syncthreads();
  aw[tid] = e * inv;
  __syncthreads();
  for (int j = tid; j < HH; j += 256) {
    float c = 0.f;
    for (int s2 = 0; s2 < SS; ++s2) c += aw[s2]*enc_out[(size_t)s2*HH + j];
    CTX[b*HH + j] = c;
  }
}

__global__ void cat_kernel(const float* __restrict__ HID2, const float* __restrict__ CTX,
                           const float* __restrict__ Wc, const float* __restrict__ bc,
                           __bf16* __restrict__ CATB)
{
  __shared__ float h2s[HH];
  __shared__ float cs[HH];
  const int b = blockIdx.x, j = threadIdx.x;       // beams x 512
  h2s[j] = HID2[b*HH + j];
  cs[j]  = CTX [b*HH + j];
  __syncthreads();
  const float* wr = Wc + (size_t)j*(2*HH);
  float s = bc[j];
  for (int k = 0; k < HH; ++k) s += h2s[k]*wr[k];
  for (int k = 0; k < HH; ++k) s += cs[k]*wr[HH + k];
  CATB[b*HH + j] = (__bf16)tanhf(s);
}

// per-beam: top-8 logits + logsumexp (so -log(softmax) = lse - logit)
__global__ void beam_topk_kernel(const float* __restrict__ LOGITS,
                                 float* __restrict__ TOPV, int* __restrict__ TOPI,
                                 float* __restrict__ LSE)
{
  __shared__ float lm[256], lsum[256];
  __shared__ float allv[256*8];
  __shared__ int   alli[256*8];
  const int b = blockIdx.x, tid = threadIdx.x;     // beams x 256
  const float* row = LOGITS + (size_t)b*VP;
  float tv[8]; int ti[8];
#pragma unroll
  for (int i = 0; i < 8; ++i) { tv[i] = -__builtin_inff(); ti[i] = 0; }
  float m = -__builtin_inff(), s = 0.f;
  for (int j = tid; j < VV; j += 256) {
    const float x = row[j];
    if (x > m) { s = s*expf(m - x) + 1.f; m = x; }
    else       { s += expf(x - m); }
    if (x > tv[7]) {
      int p = 7;
      while (p > 0 && tv[p-1] < x) { tv[p] = tv[p-1]; ti[p] = ti[p-1]; --p; }
      tv[p] = x; ti[p] = j;
    }
  }
  lm[tid] = m; lsum[tid] = s;
#pragma unroll
  for (int i = 0; i < 8; ++i) { allv[tid*8 + i] = tv[i]; alli[tid*8 + i] = ti[i]; }
  for (int off = 128; off > 0; off >>= 1) {
    __syncthreads();
    if (tid < off) {
      const float m2 = lm[tid+off], s2 = lsum[tid+off];
      const float M = fmaxf(lm[tid], m2);
      lsum[tid] = lsum[tid]*expf(lm[tid] - M) + s2*expf(m2 - M);
      lm[tid] = M;
    }
  }
  __syncthreads();
  if (tid == 0) {
    LSE[b] = lm[0] + logf(lsum[0]);
    for (int sel = 0; sel < 8; ++sel) {
      float best = -__builtin_inff(); int bi = 0;
      for (int q = 0; q < 256*8; ++q) if (allv[q] > best) { best = allv[q]; bi = q; }
      TOPV[b*8 + sel] = best; TOPI[b*8 + sel] = alli[bi];
      allv[bi] = -__builtin_inff();
    }
  }
}

// step-0: seed the 8 beams from beam 0's top-8
__global__ void init_update_kernel(const float* __restrict__ TOPV, const int* __restrict__ TOPI,
                                   const float* __restrict__ LSE, const float* __restrict__ HID2,
                                   float* __restrict__ HID, float* __restrict__ SCORES,
                                   int* __restrict__ LAST, int* __restrict__ TOKA)
{
  const int tid = threadIdx.x;                     // 512 == KB*TT
  TOKA[tid] = 0;
  __syncthreads();
  if (tid < KB) {
    SCORES[tid] = LSE[0] - TOPV[tid];              // -log(p)
    const int tk = TOPI[tid];
    LAST[tid] = tk;
    TOKA[tid*TT] = tk;
  }
  for (int b = 0; b < KB; ++b) HID[b*HH + tid] = HID2[tid];
}

// steps t>=1: merge 8x8 candidates, remap tokens / hidden by parent
__global__ void beam_update_kernel(const float* __restrict__ TOPV, const int* __restrict__ TOPI,
                                   const float* __restrict__ LSE, const float* __restrict__ HID2,
                                   float* __restrict__ HID, float* __restrict__ SCORES,
                                   int* __restrict__ LAST, int* __restrict__ TOKA,
                                   int* __restrict__ TOKB, int t)
{
  __shared__ int par[KB], tks[KB];
  __shared__ float ns[KB];
  const int tid = threadIdx.x;                     // 512
  if (tid == 0) {
    float cand[KB*8];
    for (int b = 0; b < KB; ++b)
      for (int j = 0; j < 8; ++j)
        cand[b*8 + j] = SCORES[b] * (LSE[b] - TOPV[b*8 + j]);
    for (int sel = 0; sel < KB; ++sel) {
      float best = __builtin_inff(); int bi = 0;
      for (int q = 0; q < KB*8; ++q) if (cand[q] < best) { best = cand[q]; bi = q; }
      ns[sel] = best; par[sel] = bi >> 3; tks[sel] = TOPI[bi];
      cand[bi] = __builtin_inff();
    }
  }
  __syncthreads();
  const int i = tid / TT, c = tid % TT;
  TOKB[tid] = TOKA[par[i]*TT + c];
  __syncthreads();
  if (tid < KB) {
    TOKB[tid*TT + t] = tks[tid];
    SCORES[tid] = ns[tid];
    LAST[tid]   = tks[tid];
  }
  for (int b = 0; b < KB; ++b)
    HID[b*HH + tid] = HID2[par[b]*HH + tid];
  __syncthreads();
  TOKA[tid] = TOKB[tid];
}

__global__ void finalize_kernel(const int* __restrict__ TOKA, const float* __restrict__ SCORES,
                                float* __restrict__ out)
{
  const int idx = blockIdx.x*blockDim.x + threadIdx.x;
  if (idx < KB*TT)           out[idx] = (float)TOKA[idx];
  else if (idx < KB*TT + KB) out[idx] = SCORES[idx - KB*TT];
}

// ---------------------------------------------------------------------------
extern "C" void kernel_launch(void* const* d_in, const int* in_sizes, int n_in,
                              void* d_out, int out_size, void* d_ws, size_t ws_size,
                              hipStream_t stream)
{
  (void)in_sizes; (void)n_in; (void)out_size; (void)ws_size;
  const int*   input_seq = (const int*)  d_in[0];
  const float* enc_emb   = (const float*)d_in[3];
  const float* eWih_f    = (const float*)d_in[4];
  const float* eWhh_f    = (const float*)d_in[5];
  const float* ebih_f    = (const float*)d_in[6];
  const float* ebhh_f    = (const float*)d_in[7];
  const float* eWih_b    = (const float*)d_in[8];
  const float* eWhh_b    = (const float*)d_in[9];
  const float* ebih_b    = (const float*)d_in[10];
  const float* ebhh_b    = (const float*)d_in[11];
  const float* dec_emb   = (const float*)d_in[12];
  const float* dWih      = (const float*)d_in[13];
  const float* dWhh      = (const float*)d_in[14];
  const float* dbih      = (const float*)d_in[15];
  const float* dbhh      = (const float*)d_in[16];
  const float* Wc        = (const float*)d_in[17];
  const float* bc        = (const float*)d_in[18];
  const float* Wout      = (const float*)d_in[19];
  const float* bout      = (const float*)d_in[20];
  float* out = (float*)d_out;

  // ---- workspace carve-up (256B aligned) ----
  char* w = (char*)d_ws;
  auto alloc = [&](size_t bytes) { char* p = w; w += (bytes + 255) & ~(size_t)255; return p; };
  float*  EMB    = (float*) alloc((size_t)SS*HH*4);
  float*  GXF    = (float*) alloc((size_t)SS*G3*4);
  float*  GXB    = (float*) alloc((size_t)SS*G3*4);
  float*  ENCOUT = (float*) alloc((size_t)SS*HH*4);
  float*  DECH0  = (float*) alloc((size_t)HH*4);
  float*  X      = (float*) alloc((size_t)16*HH*4);
  float*  HID    = (float*) alloc((size_t)16*HH*4);
  float*  HID2   = (float*) alloc((size_t)16*HH*4);
  float*  GI     = (float*) alloc((size_t)16*G3*4);
  float*  GH     = (float*) alloc((size_t)16*G3*4);
  float*  CTX    = (float*) alloc((size_t)16*HH*4);
  __bf16* CATB   = (__bf16*)alloc((size_t)16*HH*2);
  float*  TOPV   = (float*) alloc((size_t)KB*8*4);
  int*    TOPI   = (int*)   alloc((size_t)KB*8*4);
  float*  LSE    = (float*) alloc((size_t)KB*4);
  float*  SCORES = (float*) alloc((size_t)KB*4);
  int*    LAST   = (int*)   alloc((size_t)KB*4);
  int*    TOKA   = (int*)   alloc((size_t)KB*TT*4);
  int*    TOKB   = (int*)   alloc((size_t)KB*TT*4);
  float*  LOGITS = (float*) alloc((size_t)16*VP*4);
  __bf16* WOUTB  = (__bf16*)alloc((size_t)VP*HH*2);

  // ---- prep ----
  embed_gather_kernel<<<SS, HH, 0, stream>>>(input_seq, enc_emb, EMB);
  {
    const size_t n = (size_t)VP*HH;
    wout_convert_kernel<<<(unsigned)((n + 255)/256), 256, 0, stream>>>(Wout, WOUTB);
  }
  // input-gate precompute for both directions: S x 3H = (S x H) @ (3H x H)^T
  gemm_f32_wmma<<<dim3(G3/16, SS/16), 32, 0, stream>>>(EMB, HH, SS, 0, eWih_f, HH, ebih_f, GXF, G3);
  gemm_f32_wmma<<<dim3(G3/16, SS/16), 32, 0, stream>>>(EMB, HH, SS, 1, eWih_b, HH, ebih_b, GXB, G3);
  // sequential recurrences
  enc_dir_kernel<<<1, HH, 0, stream>>>(GXF, eWhh_f, ebhh_f, ENCOUT, DECH0, 0);
  enc_dir_kernel<<<1, HH, 0, stream>>>(GXB, eWhh_b, ebhh_b, ENCOUT, DECH0, 1);
  hid_init_kernel<<<16, HH, 0, stream>>>(DECH0, HID, HID2, CATB);

  // ---- one decoder step (beams rows live, padded to 16) ----
  auto dec_step = [&](int beams, int use_sos) {
    gather_x_kernel<<<16, HH, 0, stream>>>(dec_emb, LAST, beams, use_sos, X);
    gemm_f32_wmma<<<dim3(G3/16, 1), 32, 0, stream>>>(X,   HH, 16, 0, dWih, HH, dbih, GI, G3);
    gemm_f32_wmma<<<dim3(G3/16, 1), 32, 0, stream>>>(HID, HH, 16, 0, dWhh, HH, dbhh, GH, G3);
    gru_combine_kernel<<<beams, HH, 0, stream>>>(GI, GH, HID, HID2);
    attn_kernel<<<beams, 256, 0, stream>>>(HID2, ENCOUT, CTX);
    cat_kernel<<<beams, HH, 0, stream>>>(HID2, CTX, Wc, bc, CATB);
    gemm_bf16_wmma<<<VP/64, 128, 0, stream>>>(CATB, WOUTB, bout, VV, LOGITS, VP);
    beam_topk_kernel<<<beams, 256, 0, stream>>>(LOGITS, TOPV, TOPI, LSE);
  };

  // step 0: single beam seeded from SOS, expand to K beams
  dec_step(1, 1);
  init_update_kernel<<<1, KB*TT, 0, stream>>>(TOPV, TOPI, LSE, HID2, HID, SCORES, LAST, TOKA);

  // steps 1..T-1
  for (int t = 1; t < TT; ++t) {
    dec_step(KB, 0);
    beam_update_kernel<<<1, KB*TT, 0, stream>>>(TOPV, TOPI, LSE, HID2, HID, SCORES, LAST,
                                                TOKA, TOKB, t);
  }

  finalize_kernel<<<2, 512, 0, stream>>>(TOKA, SCORES, out);
}
